// MultiHeadedAttentionPure_66022237274782
// MI455X (gfx1250) — matrix operand order repaired and checked by reading, not compiled, not run on hardware
//
#include <hip/hip_runtime.h>

typedef __bf16 bf16;
typedef bf16  bf16x16 __attribute__((ext_vector_type(16)));
typedef bf16  bf16x8  __attribute__((ext_vector_type(8)));
typedef float f32x8   __attribute__((ext_vector_type(8)));
typedef float f32x4   __attribute__((ext_vector_type(4)));
typedef unsigned int u32x4 __attribute__((ext_vector_type(4)));
typedef int          i32x8 __attribute__((ext_vector_type(8)));
typedef int          i32x4 __attribute__((ext_vector_type(4)));

#define F_DIM 512
#define T_DIM 1024
#define B_DIM 8
#define H_DIM 8
#define D_DIM 64
#define BT_DIM (B_DIM * T_DIM)
#define MIN_VALF (-3.4028234663852886e38f)

__device__ __forceinline__ float bf2f(bf16 x) { return (float)x; }
__device__ __forceinline__ bf16  f2bf(float x) { return (bf16)x; }
__device__ __forceinline__ f32x8 zero8() { f32x8 z = {0,0,0,0,0,0,0,0}; return z; }

// v_wmma_f32_16x16x32_bf16: D = A(16x32) * B(32x16) + C(16x16)
__device__ __forceinline__ f32x8 wmma_bf16(bf16x16 a, bf16x16 b, f32x8 c) {
    return __builtin_amdgcn_wmma_f32_16x16x32_bf16(
        false, a, false, b, (short)0, c, false, false);
}

// A-fragment (16x32, 16-bit) K index for vector element e (0..15), lane-half lh.
// Per lane this is two contiguous runs: K = lh*8 + [0..7] and K = lh*8+16 + [0..7].
__device__ __forceinline__ int a_kidx(int e, int lh) {
    int i = e >> 1, j = e & 1;
    return ((i >> 2) << 4) + (lh << 3) + ((i & 3) << 1) + j;
}
// B-fragment (32x16): K = (lane>>4)*8 + (e&7) + (e>>3)*16 ; N = lane&15.
// C/D fragment (16x16 f32): element e -> M = e + 8*(lane>>4), N = lane&15.

union afrag_u { bf16x16 v; bf16x8 h[2]; };

// ---------------------------------------------------------------------------
// TDM: 2D tensor_load_to_lds (6-arg builtin on this toolchain).
// D# per ISA ch.8: group0 = count/lds_addr/global_addr/type=2;
// group1 = data_size, tensor dims, tile dims, dim0 stride.
// ---------------------------------------------------------------------------
__device__ __forceinline__ void tdm_load_2d(unsigned int lds_off, const void* gaddr,
                                            unsigned int dsize_code,
                                            unsigned int tile0, unsigned int tile1,
                                            unsigned long long stride0)
{
    const unsigned long long ga = (unsigned long long)(uintptr_t)gaddr;
    const unsigned int td0 = 1u << 20, td1 = 1u << 20;   // no OOB clipping
    u32x4 g0;
    g0.x = 1u;                                           // count=1, user mode
    g0.y = lds_off;
    g0.z = (unsigned int)ga;
    g0.w = (unsigned int)((ga >> 32) & 0x01FFFFFFu) | (2u << 30);  // type=2
    i32x8 g1;
    g1[0] = (int)(dsize_code << 16);
    g1[1] = (int)((td0 & 0xFFFFu) << 16);
    g1[2] = (int)((td0 >> 16) | ((td1 & 0xFFFFu) << 16));
    g1[3] = (int)((td1 >> 16) | (tile0 << 16));
    g1[4] = (int)tile1;
    g1[5] = (int)(unsigned int)(stride0 & 0xFFFFFFFFu);
    g1[6] = (int)(unsigned int)((stride0 >> 32) & 0xFFFFu);
    g1[7] = 0;
    i32x4 z4 = {0, 0, 0, 0};
    i32x8 z8 = {0, 0, 0, 0, 0, 0, 0, 0};
    __builtin_amdgcn_tensor_load_to_lds(g0, g1, z4, z4, z8, 0);
}

__device__ __forceinline__ unsigned int lds_offset(const void* p) {
    return (unsigned int)(uintptr_t)p;   // flat->LDS aperture: addr[31:0]
}

// ---------------------------------------------------------------------------
// Vectorized f32 -> bf16 conversion (8 elements/thread, 16B stores).
// ---------------------------------------------------------------------------
__global__ __launch_bounds__(256)
void cvt_f32_bf16(const float* __restrict__ in, bf16* __restrict__ outp, int n8)
{
    const int i = blockIdx.x * 256 + threadIdx.x;
    if (i >= n8) return;
    const f32x4* p = (const f32x4*)in + (size_t)i * 2;
    const f32x4 a = p[0], b = p[1];
    bf16x8 o;
    o[0] = f2bf(a.x); o[1] = f2bf(a.y); o[2] = f2bf(a.z); o[3] = f2bf(a.w);
    o[4] = f2bf(b.x); o[5] = f2bf(b.y); o[6] = f2bf(b.z); o[7] = f2bf(b.w);
    *((bf16x8*)outp + i) = o;
}

// ---------------------------------------------------------------------------
// Pure-bf16 GEMM: out[M x 512] = A[M x 512] * W^T + bias (W bf16 [512 x 512]).
// Block = 256 thr (8 waves) -> 128 x 64 tile.  Per K-chunk (32): TDM loads
// A-tile (128x32 bf16, 8KB) + W-tile (64x32 bf16, 4KB) into LDS, double
// buffered; fragments are contiguous 16B ds loads; no conversion in the loop.
// ---------------------------------------------------------------------------
__global__ __launch_bounds__(256)
void gemm_wmma_bf16(const bf16* __restrict__ A, const bf16* __restrict__ W,
                    const float* __restrict__ bias,
                    float* __restrict__ outF, bf16* __restrict__ outB, int M)
{
    __shared__ __align__(16) unsigned char smem[24576];
    // W tiles [2][64][32] bf16 at 0     (4 KB each)
    // A tiles [2][128][32] bf16 at 8192 (8 KB each)

    const int lane = threadIdx.x & 31;
    const int wave = threadIdx.x >> 5;
    const int ln = lane & 15, lh = lane >> 4;
    const int bm = blockIdx.x >> 3;
    const int bn = blockIdx.x & 7;
    const int n0 = bn * 64;
    const int row_w = wave * 16 + ln;

    f32x8 acc[4];
#pragma unroll
    for (int t = 0; t < 4; ++t) acc[t] = zero8();

    const bool loader = (wave == 0);
    const size_t a_row0 = (size_t)bm * 128 * F_DIM;

    if (loader) {
        tdm_load_2d(lds_offset(smem + 8192), A + a_row0,           1u, 32u, 128u, F_DIM);
        tdm_load_2d(lds_offset(smem),        W + (size_t)n0 * F_DIM, 1u, 32u, 64u,  F_DIM);
    }

    for (int kc = 0; kc < F_DIM / 32; ++kc) {
        const int cur = kc & 1;
        if (loader) {
            if (kc + 1 < F_DIM / 32) {
                const int k1 = (kc + 1) * 32;
                tdm_load_2d(lds_offset(smem + 8192 + (cur ^ 1) * 8192),
                            A + a_row0 + k1, 1u, 32u, 128u, F_DIM);
                tdm_load_2d(lds_offset(smem + (cur ^ 1) * 4096),
                            W + (size_t)n0 * F_DIM + k1, 1u, 32u, 64u, F_DIM);
                __builtin_amdgcn_s_wait_tensorcnt(2);   // current buffer done
            } else {
                __builtin_amdgcn_s_wait_tensorcnt(0);
            }
        }
        __syncthreads();

        const bf16* Wt = (const bf16*)(smem + cur * 4096);
        const bf16* At = (const bf16*)(smem + 8192 + cur * 8192);

        // A fragment: two contiguous 16B runs per lane
        afrag_u a;
        const int abase = row_w * 32 + (lh << 3);
        a.h[0] = *(const bf16x8*)(At + abase);
        a.h[1] = *(const bf16x8*)(At + abase + 16);

#pragma unroll
        for (int t = 0; t < 4; ++t) {
            afrag_u b;
            const int wbase = (t * 16 + ln) * 32 + (lh << 3);
            b.h[0] = *(const bf16x8*)(Wt + wbase);
            b.h[1] = *(const bf16x8*)(Wt + wbase + 16);
            acc[t] = wmma_bf16(a.v, b.v, acc[t]);
        }
        __syncthreads();
    }

#pragma unroll
    for (int t = 0; t < 4; ++t) {
#pragma unroll
        for (int e = 0; e < 8; ++e) {
            const int m = bm * 128 + wave * 16 + e + (lh << 3);
            const int n = n0 + t * 16 + ln;
            float v = acc[t][e] + (bias ? bias[n] : 0.0f);
            if (outF) outF[(size_t)m * F_DIM + n] = v;
            else      outB[(size_t)m * F_DIM + n] = f2bf(v);
        }
    }
}

// ---------------------------------------------------------------------------
// Fused relative-position attention (flash-style).  One wave per
// (b, h, 16-row query tile); T2 in tiles of 32; online masked softmax.
// ---------------------------------------------------------------------------
__global__ __launch_bounds__(32)
void attn_wmma(const bf16* __restrict__ Qb, const bf16* __restrict__ Kb,
               const bf16* __restrict__ Vb, const bf16* __restrict__ Pb,
               const int*  __restrict__ mask,
               const float* __restrict__ ubias, const float* __restrict__ vbias,
               bf16* __restrict__ Xb)
{
    __shared__ bf16 pls[16 * 32];

    const int lane = threadIdx.x & 31;
    const int ln = lane & 15, lh = lane >> 4;
    int blk = blockIdx.x;
    const int t0 = (blk % (T_DIM / 16)) * 16; blk /= (T_DIM / 16);
    const int h  = blk % H_DIM;               blk /= H_DIM;
    const int b  = blk;
    const float scale = 0.125f;               // 1/sqrt(64)

    bf16x16 qu[2], qv[2];
    const size_t qrow = ((size_t)(b * T_DIM + t0 + ln)) * F_DIM + h * D_DIM;
#pragma unroll
    for (int c = 0; c < 2; ++c) {
#pragma unroll
        for (int e = 0; e < 16; ++e) {
            const int d = c * 32 + a_kidx(e, lh);
            const float q = bf2f(Qb[qrow + d]);
            qu[c][e] = f2bf(q + ubias[h * D_DIM + d]);
            qv[c][e] = f2bf(q + vbias[h * D_DIM + d]);
        }
    }

    f32x8 oacc[4];
#pragma unroll
    for (int t = 0; t < 4; ++t) oacc[t] = zero8();
    float mstate[8], lstate[8];
#pragma unroll
    for (int e = 0; e < 8; ++e) { mstate[e] = -INFINITY; lstate[e] = 0.0f; }

    for (int s0 = 0; s0 < T_DIM; s0 += 32) {
        f32x8 sc[2];
#pragma unroll
        for (int half = 0; half < 2; ++half) {
            f32x8 acc = zero8();
            const int s = s0 + half * 16 + ln;
            const size_t krow = ((size_t)(b * T_DIM + s)) * F_DIM + h * D_DIM;
            const size_t prow = ((size_t)s) * F_DIM + h * D_DIM;
#pragma unroll
            for (int c = 0; c < 2; ++c) {
                afrag_u kf, pf;
                const int dbase = c * 32 + (lh << 3);
                kf.h[0] = *(const bf16x8*)(Kb + krow + dbase);
                kf.h[1] = *(const bf16x8*)(Kb + krow + dbase + 16);
                pf.h[0] = *(const bf16x8*)(Pb + prow + dbase);
                pf.h[1] = *(const bf16x8*)(Pb + prow + dbase + 16);
                acc = wmma_bf16(qu[c], kf.v, acc);        // matrix_ac
                acc = wmma_bf16(qv[c], pf.v, acc);        // matrix_bd
            }
            sc[half] = acc;
        }

        int mb0[8], mb1[8];
#pragma unroll
        for (int e = 0; e < 8; ++e) {
            const int t = t0 + e + (lh << 3);
            const int* mrow = mask + ((size_t)b * T_DIM + t) * T_DIM;
            mb0[e] = mrow[s0 + ln];
            mb1[e] = mrow[s0 + 16 + ln];
            sc[0][e] = mb0[e] ? sc[0][e] * scale : MIN_VALF;
            sc[1][e] = mb1[e] ? sc[1][e] * scale : MIN_VALF;
        }

        float corr[8];
#pragma unroll
        for (int e = 0; e < 8; ++e) {
            float rm = fmaxf(sc[0][e], sc[1][e]);
#pragma unroll
            for (int off = 1; off < 16; off <<= 1)
                rm = fmaxf(rm, __shfl_xor(rm, off, 32));
            const float nm = fmaxf(mstate[e], rm);
            const float c0 = __expf(mstate[e] - nm);
            const float p0 = __expf(sc[0][e] - nm);
            const float p1 = __expf(sc[1][e] - nm);
            float rs = p0 + p1;
#pragma unroll
            for (int off = 1; off < 16; off <<= 1)
                rs += __shfl_xor(rs, off, 32);
            lstate[e] = lstate[e] * c0 + rs;
            mstate[e] = nm;
            corr[e] = c0;
            const int row = e + (lh << 3);
            pls[row * 32 + ln]      = f2bf(mb0[e] ? p0 : 0.0f);
            pls[row * 32 + 16 + ln] = f2bf(mb1[e] ? p1 : 0.0f);
        }
        __syncthreads();

        afrag_u pa;
        pa.h[0] = *(const bf16x8*)(pls + ln * 32 + (lh << 3));
        pa.h[1] = *(const bf16x8*)(pls + ln * 32 + (lh << 3) + 16);
        __syncthreads();

#pragma unroll
        for (int oc = 0; oc < 4; ++oc) {
#pragma unroll
            for (int e = 0; e < 8; ++e) oacc[oc][e] *= corr[e];
            bf16x16 vf;
#pragma unroll
            for (int e = 0; e < 16; ++e) {
                const int ks = (lh << 3) + (e & 7) + ((e >> 3) << 4);
                vf[e] = Vb[((size_t)(b * T_DIM + s0 + ks)) * F_DIM
                           + h * D_DIM + oc * 16 + ln];
            }
            oacc[oc] = wmma_bf16(pa.v, vf, oacc[oc]);
        }
    }

#pragma unroll
    for (int oc = 0; oc < 4; ++oc) {
#pragma unroll
        for (int e = 0; e < 8; ++e) {
            const int t = t0 + e + (lh << 3);
            const float r = oacc[oc][e] / lstate[e];
            Xb[((size_t)(b * T_DIM + t)) * F_DIM + h * D_DIM + oc * 16 + ln] = f2bf(r);
        }
    }
}

// ---------------------------------------------------------------------------
extern "C" void kernel_launch(void* const* d_in, const int* in_sizes, int n_in,
                              void* d_out, int out_size, void* d_ws, size_t ws_size,
                              hipStream_t stream) {
    const float* query   = (const float*)d_in[0];
    const float* key     = (const float*)d_in[1];
    const float* value   = (const float*)d_in[2];
    const int*   mask    = (const int*)  d_in[3];
    const float* pos_emb = (const float*)d_in[4];
    const float* Wq = (const float*)d_in[5];  const float* bq = (const float*)d_in[6];
    const float* Wk = (const float*)d_in[7];  const float* bk = (const float*)d_in[8];
    const float* Wv = (const float*)d_in[9];  const float* bv = (const float*)d_in[10];
    const float* Wp = (const float*)d_in[11];
    const float* Wo = (const float*)d_in[12]; const float* bo = (const float*)d_in[13];
    const float* pbu = (const float*)d_in[14];
    const float* pbv = (const float*)d_in[15];
    float* out = (float*)d_out;

    const size_t ACT = (size_t)BT_DIM * F_DIM;   // 4.19M elems
    const size_t POS = (size_t)T_DIM * F_DIM;    // 0.52M
    const size_t WEL = (size_t)F_DIM * F_DIM;    // 0.26M

    bf16* Qb  = (bf16*)d_ws;          // attention operands (bf16)
    bf16* Kb  = Qb + ACT;
    bf16* Vb  = Kb + ACT;
    bf16* Pb  = Vb + ACT;
    bf16* Xb  = Pb + POS;
    bf16* Qc  = Xb + ACT;             // bf16 copies of f32 inputs
    bf16* Kc  = Qc + ACT;
    bf16* Vc  = Kc + ACT;
    bf16* Pc  = Vc + ACT;
    bf16* Wqc = Pc + POS;
    bf16* Wkc = Wqc + WEL;
    bf16* Wvc = Wkc + WEL;
    bf16* Wpc = Wvc + WEL;
    bf16* Woc = Wpc + WEL;            // total ~63 MB

    // f32 -> bf16 conversions (bandwidth-trivial: ~58 MB read + ~29 MB written)
    cvt_f32_bf16<<<(int)(ACT / 8 / 256), 256, 0, stream>>>(query,   Qc,  (int)(ACT / 8));
    cvt_f32_bf16<<<(int)(ACT / 8 / 256), 256, 0, stream>>>(key,     Kc,  (int)(ACT / 8));
    cvt_f32_bf16<<<(int)(ACT / 8 / 256), 256, 0, stream>>>(value,   Vc,  (int)(ACT / 8));
    cvt_f32_bf16<<<(int)(POS / 8 / 256), 256, 0, stream>>>(pos_emb, Pc,  (int)(POS / 8));
    cvt_f32_bf16<<<(int)(WEL / 8 / 256), 256, 0, stream>>>(Wq,      Wqc, (int)(WEL / 8));
    cvt_f32_bf16<<<(int)(WEL / 8 / 256), 256, 0, stream>>>(Wk,      Wkc, (int)(WEL / 8));
    cvt_f32_bf16<<<(int)(WEL / 8 / 256), 256, 0, stream>>>(Wv,      Wvc, (int)(WEL / 8));
    cvt_f32_bf16<<<(int)(WEL / 8 / 256), 256, 0, stream>>>(Wp,      Wpc, (int)(WEL / 8));
    cvt_f32_bf16<<<(int)(WEL / 8 / 256), 256, 0, stream>>>(Wo,      Woc, (int)(WEL / 8));

    const int blocksA = (BT_DIM / 128) * (F_DIM / 64);  // 512
    const int blocksP = (T_DIM / 128) * (F_DIM / 64);   // 64

    gemm_wmma_bf16<<<blocksA, 256, 0, stream>>>(Qc, Wqc, bq,      nullptr, Qb, BT_DIM);
    gemm_wmma_bf16<<<blocksA, 256, 0, stream>>>(Kc, Wkc, bk,      nullptr, Kb, BT_DIM);
    gemm_wmma_bf16<<<blocksA, 256, 0, stream>>>(Vc, Wvc, bv,      nullptr, Vb, BT_DIM);
    gemm_wmma_bf16<<<blocksP, 256, 0, stream>>>(Pc, Wpc, nullptr, nullptr, Pb, T_DIM);

    attn_wmma<<<B_DIM * H_DIM * (T_DIM / 16), 32, 0, stream>>>(
        Qb, Kb, Vb, Pb, mask, pbu, pbv, Xb);

    gemm_wmma_bf16<<<blocksA, 256, 0, stream>>>(Xb, Woc, bo, out, nullptr, BT_DIM);
}
